// WholeCell_19602230739411
// MI455X (gfx1250) — matrix operand order repaired and checked by reading, not compiled, not run on hardware
//
#include <hip/hip_runtime.h>
#include <hip/hip_bf16.h>

// Problem constants (match reference module constants)
#define NN 10000
#define BB 32
#define DD 32
#define HH 100
#define HP 112    // H padded to multiple of 16 (rows / g dim)
#define KP2 128   // layer-2 K (h) dim padded to multiple of 32 in W2 workspace
#define TT 8
#define WPB 4     // waves (nodes) per block

typedef unsigned short u16;
typedef __attribute__((ext_vector_type(8)))  u16    v8u;
typedef __attribute__((ext_vector_type(16))) u16    v16u;
typedef __attribute__((ext_vector_type(16))) __bf16 v16bf;
typedef __attribute__((ext_vector_type(8)))  float  v8f;

__device__ __forceinline__ u16 f2bf(float f) {
  unsigned u = __builtin_bit_cast(unsigned, f);
  unsigned r = 0x7FFFu + ((u >> 16) & 1u);   // round-to-nearest-even
  return (u16)((u + r) >> 16);
}

__device__ __forceinline__ float lrelu(float v) {
  return v > 0.0f ? v : 0.01f * v;
}

// A-fragment from fragment-major LDS staging: each lane's 16 halves are
// contiguous 32 bytes -> single v16u load (2 x ds_load_b128).
__device__ __forceinline__ v16bf fragAc(const u16* __restrict__ p) {
  return __builtin_bit_cast(v16bf, *(const v16u*)p);
}

// B-fragment (32x16 bf16), unconditional. Source is W stored [ncol][k]
// row-major (K padded in workspace), so column n of B is row n of W.
// CDNA5 16-bit B layout: lanes 0-15 col N=lane K=0..7,16..23; lanes 16-31 col
// N=lane-16 K=8..15,24..31 -> two aligned 16-byte global loads per lane.
__device__ __forceinline__ v16bf fragB(const u16* __restrict__ base, int stride,
                                       int lane) {
  const int half = lane >> 4;
  const int n    = lane & 15;
  const v8u lo = *(const v8u*)(base + n * stride + half * 8);
  const v8u hi = *(const v8u*)(base + n * stride + 16 + half * 8);
  const v16u v = __builtin_shufflevector(lo, hi, 0, 1, 2, 3, 4, 5, 6, 7, 8, 9, 10, 11, 12, 13, 14, 15);
  return __builtin_bit_cast(v16bf, v);
}

// Swizzle helpers for fragment-major A staging.
// Element with K-index k lives at: lane' += (k&8)<<1 ; pos = ((k>>4)<<3)|(k&7)
__device__ __forceinline__ int a_lane_add(int k) { return (k & 8) << 1; }
__device__ __forceinline__ int a_pos(int k)      { return ((k >> 4) << 3) | (k & 7); }

// One-time weight conversion: f32 -> bf16, zero-padded.
__global__ void convert_w1(const float* __restrict__ W1, u16* __restrict__ W1w) {
  int i = blockIdx.x * blockDim.x + threadIdx.x;     // NN*HP*DD
  if (i >= NN * HP * DD) return;
  int d  = i % DD;
  int hp = (i / DD) % HP;
  int n  = i / (DD * HP);
  float v = (hp < HH) ? W1[(n * HH + hp) * DD + d] : 0.0f;
  W1w[i] = f2bf(v);
}

__global__ void convert_w2(const float* __restrict__ W2, u16* __restrict__ W2w) {
  int i = blockIdx.x * blockDim.x + threadIdx.x;     // NN*HP*KP2
  if (i >= NN * HP * KP2) return;
  int h = i % KP2;
  int g = (i / KP2) % HP;
  int n = i / (KP2 * HP);
  float v = (g < HH && h < HH) ? W2[(n * HH + g) * HH + h] : 0.0f;
  W2w[i] = f2bf(v);
}

// One MLP iteration. One wave per node; WPB waves per block.
__launch_bounds__(WPB * 32)
__global__ void mlp_iter(const float* __restrict__ s_in,
                         float* __restrict__ s_out,
                         const u16* __restrict__ W1w,
                         const u16* __restrict__ W2w,
                         const float* __restrict__ b1,
                         const float* __restrict__ b2,
                         const float* __restrict__ W3,
                         const int* __restrict__ pidx) {
  // Fragment-major staging: [mt][lane][16] and [mt][kt][lane][16] halves.
  __shared__ __align__(32) u16   lds_x[WPB][2 * 32 * 16];       // 2 KB / wave
  __shared__ __align__(32) u16   lds_h[WPB][2 * 4 * 32 * 16];   // 8 KB / wave
  __shared__ __align__(16) float lds_r[WPB][32 * 8];            // 1 KB / wave

  const int lane = threadIdx.x & 31;
  const int wave = threadIdx.x >> 5;
  const int node = blockIdx.x * WPB + wave;

  u16*   xf = lds_x[wave];
  u16*   hf = lds_h[wave];
  float* rb = lds_r[wave];

  const u16*   w1n = W1w + (size_t)node * HP * DD;
  const u16*   w2n = W2w + (size_t)node * HP * KP2;
  const float* b1n = b1 + node * HH;
  const float* b2n = b2 + node * HH;
  const float* w3n = W3 + node * HH;

  __builtin_prefetch(w1n, 0, 0);
  __builtin_prefetch(w2n, 0, 0);

  // Pre-zero the K-pad half (pos 8..15, i.e. h=112..127) of kt==3 A fragments.
  {
    const v8u z = {0, 0, 0, 0, 0, 0, 0, 0};
#pragma unroll
    for (int mt = 0; mt < 2; ++mt)
      *(v8u*)(hf + ((mt * 4 + 3) * 32 + lane) * 16 + 8) = z;
  }

  // ---- Gather x[b][d] = s_in[b][pred_idx[node][d]], bf16, fragment-major.
  {
    const int d    = lane;                 // lane == d
    const int ladd = a_lane_add(d);
    const int pos  = a_pos(d);
    const int idx  = pidx[node * DD + d];
#pragma unroll 8
    for (int b = 0; b < BB; ++b) {
      float v = s_in[b * NN + idx];
      xf[(((b >> 4) * 32) + (b & 15) + ladd) * 16 + pos] = f2bf(v);
    }
  }

  // ---- Layer 1: (32x32) x (32x112) -> LDS h (bias + leaky-relu, bf16)
#pragma unroll
  for (int mt = 0; mt < 2; ++mt) {
    const v16bf a = fragAc(xf + (mt * 32 + lane) * 16);
#pragma unroll
    for (int ht = 0; ht < 7; ++ht) {
      const v16bf bm = fragB(w1n + ht * 16 * DD, DD, lane);
      v8f acc = {};
      acc = __builtin_amdgcn_wmma_f32_16x16x32_bf16(false, a, false, bm,
                                                    (short)0, acc, false, false);
      // Epilogue: write into fragment-major layout for layer-2 A operand.
      const int col   = ht * 16 + (lane & 15);     // h index (K of layer 2)
      const int cHalf = lane >> 4;                 // C layout: rows cHalf*8 + r
      const bool live = (ht < 6) || (col < HH);    // compile-time for ht<6
      const float bias = live ? b1n[col] : 0.0f;
      const int kt  = col >> 5;
      const int k   = col & 31;
      u16* dst0 = hf + ((mt * 4 + kt) * 32 + cHalf * 8 + a_lane_add(k)) * 16
                     + a_pos(k);
#pragma unroll
      for (int r = 0; r < 8; ++r) {
        u16 val = f2bf(lrelu(acc[r] + bias));
        if (ht == 6 && !live) val = 0;             // only tile 6 straddles pad
        dst0[r * 16] = val;
      }
    }
  }

  // ---- Layer 2 (32x112 x 128x112, K padded) fused with Layer-3 reduction.
#pragma unroll
  for (int mt = 0; mt < 2; ++mt) {
    // Hoist the 4 A fragments (reused by all 7 gt tiles).
    const v16bf a0 = fragAc(hf + ((mt * 4 + 0) * 32 + lane) * 16);
    const v16bf a1 = fragAc(hf + ((mt * 4 + 1) * 32 + lane) * 16);
    const v16bf a2 = fragAc(hf + ((mt * 4 + 2) * 32 + lane) * 16);
    const v16bf a3 = fragAc(hf + ((mt * 4 + 3) * 32 + lane) * 16);

    float part[8];
#pragma unroll
    for (int r = 0; r < 8; ++r) part[r] = 0.0f;

#pragma unroll
    for (int gt = 0; gt < 7; ++gt) {
      const u16* wg = w2n + gt * 16 * KP2;
      v8f acc = {};   // D->C accumulate chain: designed WMMA pattern, no hazard
      acc = __builtin_amdgcn_wmma_f32_16x16x32_bf16(
          false, a0, false, fragB(wg + 0 * 32, KP2, lane), (short)0, acc, false, false);
      acc = __builtin_amdgcn_wmma_f32_16x16x32_bf16(
          false, a1, false, fragB(wg + 1 * 32, KP2, lane), (short)0, acc, false, false);
      acc = __builtin_amdgcn_wmma_f32_16x16x32_bf16(
          false, a2, false, fragB(wg + 2 * 32, KP2, lane), (short)0, acc, false, false);
      acc = __builtin_amdgcn_wmma_f32_16x16x32_bf16(
          false, a3, false, fragB(wg + 3 * 32, KP2, lane), (short)0, acc, false, false);

      const int g = gt * 16 + (lane & 15);
      const bool live = (gt < 6) || (g < HH);      // compile-time for gt<6
      const float bias = live ? b2n[g] : 0.0f;
      const float w3   = live ? w3n[g] : 0.0f;
#pragma unroll
      for (int r = 0; r < 8; ++r)
        part[r] += lrelu(acc[r] + bias) * w3;      // L3 partial per lane-column
    }

    // Cross-lane reduce the 16 g-columns per row via LDS (wave-local,
    // in-order DS pipeline; uniform addressing, single store per lane).
#pragma unroll
    for (int r = 0; r < 8; ++r) rb[lane * 8 + r] = part[r];
    if (lane < 16) {
      const int r   = lane & 7;
      const int grp = lane >> 3;                   // 0: rows 0..7 (lanes 0-15)
      float sum = 0.0f;                            // 1: rows 8..15 (lanes 16-31)
#pragma unroll
      for (int j = 0; j < 16; ++j) sum += rb[(grp * 16 + j) * 8 + r];
      const int b = mt * 16 + lane;
      s_out[b * NN + node] = lrelu(sum);
    }
  }
}

extern "C" void kernel_launch(void* const* d_in, const int* in_sizes, int n_in,
                              void* d_out, int out_size, void* d_ws, size_t ws_size,
                              hipStream_t stream) {
  const float* state = (const float*)d_in[0];
  const float* W1    = (const float*)d_in[1];
  const float* b1    = (const float*)d_in[2];
  const float* W2    = (const float*)d_in[3];
  const float* b2    = (const float*)d_in[4];
  const float* W3    = (const float*)d_in[5];
  const int*   pidx  = (const int*)d_in[6];
  // d_in[7] is T (device scalar); reference module constant T_ITERS == 8.

  char* ws = (char*)d_ws;
  size_t o = 0;
  u16* W1w = (u16*)(ws + o); o += (size_t)NN * HP * DD * sizeof(u16);  o = (o + 255) & ~(size_t)255;
  u16* W2w = (u16*)(ws + o); o += (size_t)NN * HP * KP2 * sizeof(u16); o = (o + 255) & ~(size_t)255;
  float* sA = (float*)(ws + o); o += (size_t)BB * NN * sizeof(float);  o = (o + 255) & ~(size_t)255;
  float* sB = (float*)(ws + o);

  {
    int total = NN * HP * DD;
    convert_w1<<<(total + 255) / 256, 256, 0, stream>>>(W1, W1w);
  }
  {
    int total = NN * HP * KP2;
    convert_w2<<<(total + 255) / 256, 256, 0, stream>>>(W2, W2w);
  }

  const float* cur = state;
  for (int t = 0; t < TT; ++t) {
    float* nxt = (t == TT - 1) ? (float*)d_out : ((t & 1) ? sB : sA);
    mlp_iter<<<NN / WPB, WPB * 32, 0, stream>>>(cur, nxt, W1w, W2w, b1, b2, W3, pidx);
    cur = nxt;
  }
}
